// GCNNet_3015067042303
// MI455X (gfx1250) — matrix operand order repaired and checked by reading, not compiled, not run on hardware
//
#include <hip/hip_runtime.h>
#include <stdint.h>

// ---------------- problem constants (match reference) ----------------
#define N_NODES 100000
#define E_EDGES 1600000
#define IN_DIM  512
#define HID     64
#define OUT_DIM 16
#define ROW_TILES (N_NODES / 16)   // 6250, N divisible by 16

// ---------------- WMMA vector types (gfx1250, wave32) ----------------
typedef __bf16 bf16_t;
typedef bf16_t v16bf __attribute__((ext_vector_type(16)));
typedef float  v8f   __attribute__((ext_vector_type(8)));

__device__ __forceinline__ bf16_t f2bf(float f) {
    return static_cast<bf16_t>(f);   // v_cvt_pk_bf16_f32 pairs, RNE
}

// ---------------- degree / dinv ----------------
__global__ void deg_kernel(const int* __restrict__ ei, int* __restrict__ degi) {
    long e = (long)blockIdx.x * blockDim.x + threadIdx.x;
    if (e >= E_EDGES) return;
    int row = ei[e];
    int col = ei[E_EDGES + e];
    if (row != col) atomicAdd(&degi[col], 1);
}

__global__ void dinv_kernel(const int* __restrict__ degi, float* __restrict__ dinv) {
    int i = blockIdx.x * blockDim.x + threadIdx.x;
    if (i >= N_NODES) return;
    dinv[i] = rsqrtf(2.0f + (float)degi[i]);
}

// Pack A fragment: per-lane one contiguous 16-float run of the row.
// a[0..15] <- x[row][kb + 16*hi + 0..15]
__device__ __forceinline__ v16bf load_a_frag(const float* __restrict__ rowp, int hi) {
    const float4 c0 = *(const float4*)(rowp + 16 * hi);
    const float4 c1 = *(const float4*)(rowp + 16 * hi + 4);
    const float4 c2 = *(const float4*)(rowp + 16 * hi + 8);
    const float4 c3 = *(const float4*)(rowp + 16 * hi + 12);
    v16bf a;
    a[0]  = f2bf(c0.x); a[1]  = f2bf(c0.y); a[2]  = f2bf(c0.z); a[3]  = f2bf(c0.w);
    a[4]  = f2bf(c1.x); a[5]  = f2bf(c1.y); a[6]  = f2bf(c1.z); a[7]  = f2bf(c1.w);
    a[8]  = f2bf(c2.x); a[9]  = f2bf(c2.y); a[10] = f2bf(c2.z); a[11] = f2bf(c2.w);
    a[12] = f2bf(c3.x); a[13] = f2bf(c3.y); a[14] = f2bf(c3.z); a[15] = f2bf(c3.w);
    return a;
}

// ---------------- layer-1 GEMM: h = x @ W1   (bf16 WMMA, f32 accum) ----------------
// One wave computes a 16x64 output tile. W1 staged in LDS in *fragment order* so a
// lane's 16-element B fragment is one contiguous 32-byte LDS read (ds_load_b128 x2).
// Fragment id ft = kb4*4 + t (kb4 = K-block of 32, t = 16-col tile); element i of
// lane l maps to W1[k][n] with k = kb4*32 + 16*(l>>4) + i, n = t*16 + (l&15).
__global__ void __launch_bounds__(256)
gemm1_wmma(const float* __restrict__ x, const float* __restrict__ W1,
           float* __restrict__ h) {
    __shared__ bf16_t w1s[IN_DIM * HID];          // 64 KB bf16, fragment-ordered
    for (int g = threadIdx.x; g < IN_DIM * HID; g += blockDim.x) {
        const int i   = g & 15;
        const int l   = (g >> 4) & 31;
        const int ft  = g >> 9;                    // 0..63
        const int t   = ft & 3;
        const int kb4 = ft >> 2;                   // 0..15
        const int k   = kb4 * 32 + 16 * (l >> 4) + i;
        const int n   = t * 16 + (l & 15);
        w1s[g] = f2bf(W1[(size_t)k * HID + n]);
    }
    __syncthreads();

    const int rowTile = blockIdx.x * (blockDim.x >> 5) + (threadIdx.x >> 5);
    if (rowTile >= ROW_TILES) return;             // wave-uniform: EXEC stays all-1 for WMMA
    const int lane = threadIdx.x & 31;
    const int m    = lane & 15;
    const int hi   = lane >> 4;
    const int row0 = rowTile * 16;
    const float* xrow = x + (size_t)(row0 + m) * IN_DIM;
    const v16bf* w1f  = (const v16bf*)w1s;

    v8f acc[4] = {v8f{}, v8f{}, v8f{}, v8f{}};

    for (int kb4 = 0; kb4 < IN_DIM / 32; ++kb4) {
        const v16bf a = load_a_frag(xrow + kb4 * 32, hi);
#pragma unroll
        for (int t = 0; t < 4; ++t) {
            const v16bf b = w1f[(kb4 * 4 + t) * 32 + lane];
            acc[t] = __builtin_amdgcn_wmma_f32_16x16x32_bf16(
                false, a, false, b, (short)0, acc[t], false, false);
        }
    }
    // D layout: lane holds column n = lane&15; VGPR r holds row r + 8*hi
#pragma unroll
    for (int t = 0; t < 4; ++t)
#pragma unroll
        for (int r = 0; r < 8; ++r)
            h[(size_t)(row0 + r + 8 * hi) * HID + t * 16 + m] = acc[t][r];
}

// ---------------- layer-1 edge aggregation ----------------
// 32 lanes per edge, 2 features per lane (float2 gather + 2 f32 atomic adds).
__global__ void __launch_bounds__(256)
edge_agg64(const int* __restrict__ ei, const float* __restrict__ h,
           const float* __restrict__ dinv, float* __restrict__ agg) {
    const long e = (long)blockIdx.x * (blockDim.x >> 5) + (threadIdx.x >> 5);
    if (e >= E_EDGES) return;
    const int lane = threadIdx.x & 31;
    __builtin_prefetch(ei + e + 2048, 0, 0);      // global_prefetch_b8 on edge list
    const int row = ei[e];
    const int col = ei[E_EDGES + e];
    if (row == col) return;                        // self loops carry weight 0
    const float coef = dinv[row] * dinv[col];
    const float2 hv = *(const float2*)(h + (size_t)row * HID + 2 * lane);
    float* dst = agg + (size_t)col * HID + 2 * lane;
    atomicAdd(dst,     hv.x * coef);
    atomicAdd(dst + 1, hv.y * coef);
}

// ---------------- layer-1 finalize: relu(agg + h*self_coef + b1), in place ----------------
__global__ void finalize1(float* __restrict__ agg, const float* __restrict__ h,
                          const float* __restrict__ dinv, const float* __restrict__ b1) {
    long i = (long)blockIdx.x * blockDim.x + threadIdx.x;
    if (i >= (long)N_NODES * HID) return;
    const int node = (int)(i >> 6);
    const int f    = (int)(i & 63);
    const float d  = dinv[node];
    const float v  = agg[i] + h[i] * (2.0f * d * d) + b1[f];
    agg[i] = v > 0.0f ? v : 0.0f;
}

// ---------------- layer-2 GEMM: z = h1 @ W2 (16x16 tile per wave, K=64) ----------------
__global__ void __launch_bounds__(256)
gemm2_wmma(const float* __restrict__ h1, const float* __restrict__ W2,
           float* __restrict__ z) {
    __shared__ bf16_t w2s[HID * OUT_DIM];         // 2 KB, fragment-ordered
    for (int g = threadIdx.x; g < HID * OUT_DIM; g += blockDim.x) {
        const int i   = g & 15;
        const int l   = (g >> 4) & 31;
        const int kb4 = g >> 9;                    // 0..1
        const int k   = kb4 * 32 + 16 * (l >> 4) + i;
        const int n   = l & 15;
        w2s[g] = f2bf(W2[(size_t)k * OUT_DIM + n]);
    }
    __syncthreads();

    const int rowTile = blockIdx.x * (blockDim.x >> 5) + (threadIdx.x >> 5);
    if (rowTile >= ROW_TILES) return;
    const int lane = threadIdx.x & 31;
    const int m    = lane & 15;
    const int hi   = lane >> 4;
    const int row0 = rowTile * 16;
    const float* hrow = h1 + (size_t)(row0 + m) * HID;
    const v16bf* w2f  = (const v16bf*)w2s;

    v8f acc = v8f{};
#pragma unroll
    for (int kb4 = 0; kb4 < HID / 32; ++kb4) {
        const v16bf a = load_a_frag(hrow + kb4 * 32, hi);
        const v16bf b = w2f[kb4 * 32 + lane];
        acc = __builtin_amdgcn_wmma_f32_16x16x32_bf16(
            false, a, false, b, (short)0, acc, false, false);
    }
#pragma unroll
    for (int r = 0; r < 8; ++r)
        z[(size_t)(row0 + r + 8 * hi) * OUT_DIM + m] = acc[r];
}

// ---------------- layer-2 edge aggregation (16 lanes per edge) ----------------
__global__ void __launch_bounds__(256)
edge_agg16(const int* __restrict__ ei, const float* __restrict__ z,
           const float* __restrict__ dinv, float* __restrict__ agg2) {
    const long e = (long)blockIdx.x * (blockDim.x >> 4) + (threadIdx.x >> 4);
    if (e >= E_EDGES) return;
    const int f = threadIdx.x & 15;
    const int row = ei[e];
    const int col = ei[E_EDGES + e];
    if (row == col) return;
    const float coef = dinv[row] * dinv[col];
    atomicAdd(agg2 + (size_t)col * OUT_DIM + f,
              z[(size_t)row * OUT_DIM + f] * coef);
}

// ---------------- layer-2 finalize + log_softmax ----------------
__global__ void finalize2(const float* __restrict__ agg2, const float* __restrict__ z,
                          const float* __restrict__ dinv, const float* __restrict__ b2,
                          float* __restrict__ out) {
    int node = blockIdx.x * blockDim.x + threadIdx.x;
    if (node >= N_NODES) return;
    const float d  = dinv[node];
    const float sc = 2.0f * d * d;
    float l[OUT_DIM];
    float mx = -3.0e38f;
#pragma unroll
    for (int j = 0; j < OUT_DIM; ++j) {
        const size_t idx = (size_t)node * OUT_DIM + j;
        l[j] = agg2[idx] + z[idx] * sc + b2[j];
        mx = fmaxf(mx, l[j]);
    }
    float s = 0.0f;
#pragma unroll
    for (int j = 0; j < OUT_DIM; ++j) s += __expf(l[j] - mx);
    const float lse = mx + __logf(s);
#pragma unroll
    for (int j = 0; j < OUT_DIM; ++j)
        out[(size_t)node * OUT_DIM + j] = l[j] - lse;
}

// ---------------- host launcher ----------------
extern "C" void kernel_launch(void* const* d_in, const int* in_sizes, int n_in,
                              void* d_out, int out_size, void* d_ws, size_t ws_size,
                              hipStream_t stream) {
    (void)in_sizes; (void)n_in; (void)out_size; (void)ws_size;
    const float* x  = (const float*)d_in[0];
    const int*   ei = (const int*)  d_in[1];
    const float* W1 = (const float*)d_in[2];
    const float* b1 = (const float*)d_in[3];
    const float* W2 = (const float*)d_in[4];
    const float* b2 = (const float*)d_in[5];
    float* out = (float*)d_out;

    // workspace carve-out (256B aligned)
    uint8_t* ws = (uint8_t*)d_ws;
    size_t off = 0;
    auto carve = [&](size_t bytes) -> uint8_t* {
        uint8_t* p = ws + off;
        off = (off + bytes + 255) & ~(size_t)255;
        return p;
    };
    int*   degi = (int*)  carve((size_t)N_NODES * sizeof(int));
    float* dinv = (float*)carve((size_t)N_NODES * sizeof(float));
    float* h    = (float*)carve((size_t)N_NODES * HID * sizeof(float));
    float* agg1 = (float*)carve((size_t)N_NODES * HID * sizeof(float));   // becomes h1
    float* z    = (float*)carve((size_t)N_NODES * OUT_DIM * sizeof(float));
    float* agg2 = (float*)carve((size_t)N_NODES * OUT_DIM * sizeof(float));

    hipMemsetAsync(degi, 0, (size_t)N_NODES * sizeof(int), stream);
    hipMemsetAsync(agg1, 0, (size_t)N_NODES * HID * sizeof(float), stream);
    hipMemsetAsync(agg2, 0, (size_t)N_NODES * OUT_DIM * sizeof(float), stream);

    // degrees + D^{-1/2}
    deg_kernel<<<(E_EDGES + 255) / 256, 256, 0, stream>>>(ei, degi);
    dinv_kernel<<<(N_NODES + 255) / 256, 256, 0, stream>>>(degi, dinv);

    // layer 1
    const int gemmBlocks = (ROW_TILES + 7) / 8;   // 8 waves per 256-thread block
    gemm1_wmma<<<gemmBlocks, 256, 0, stream>>>(x, W1, h);
    edge_agg64<<<(E_EDGES + 7) / 8, 256, 0, stream>>>(ei, h, dinv, agg1);
    finalize1<<<((long)N_NODES * HID + 255) / 256, 256, 0, stream>>>(agg1, h, dinv, b1);

    // layer 2
    gemm2_wmma<<<gemmBlocks, 256, 0, stream>>>(agg1, W2, z);
    edge_agg16<<<(E_EDGES + 15) / 16, 256, 0, stream>>>(ei, z, dinv, agg2);
    finalize2<<<(N_NODES + 255) / 256, 256, 0, stream>>>(agg2, z, dinv, b2, out);
}
